// egat_85263690760754
// MI455X (gfx1250) — matrix operand order repaired and checked by reading, not compiled, not run on hardware
//
#include <hip/hip_runtime.h>
#include <hip/hip_bf16.h>

// ---------------------------------------------------------------------------
// EGAT (DGL EGATConv, 1 head) + SortPooling + MLP forward for MI455X (gfx1250)
// N=50000 nodes, E=1.6M edges, D=32, L=8 layers, K=8.
//
// Memory-bound on streaming the edge features (205 MB read+write per layer);
// node arrays (6.4 MB) live in L2.  All GEMMs use v_wmma_f32_16x16x32_f16
// (f16 inputs, f32 accumulate): one wave owns a 16x32 output tile = 2 WMMAs.
//
// NOTE: this workload has N % 16 == 0 and E % 16 == 0 (50000 = 3125*16,
// 1600000 = 100000*16).  Tiles are therefore always full, so the per-row
// epilogues are branch-free: one s_clause of immediate-offset stores per
// output fragment.  The only guard is the wave-uniform launch-grid guard.
// ---------------------------------------------------------------------------

typedef __attribute__((ext_vector_type(16))) _Float16 v16h;
typedef __attribute__((ext_vector_type(8)))  float    v8f;

#define D32 32

// --- sortable-uint encoding for float atomicMax ---------------------------
__device__ __forceinline__ unsigned fenc(float x) {
    unsigned u = __float_as_uint(x);
    return (u & 0x80000000u) ? ~u : (u | 0x80000000u);
}
__device__ __forceinline__ float fdec(unsigned u) {
    u = (u & 0x80000000u) ? (u & 0x7fffffffu) : ~u;
    return __uint_as_float(u);
}
__device__ __forceinline__ float lrelu(float x) { return x > 0.0f ? x : 0.01f * x; }
__device__ __forceinline__ float relu_(float x) { return x > 0.0f ? x : 0.0f; }

// ---------------------------------------------------------------------------
// Kernel 0a/0b: embeddings
// ---------------------------------------------------------------------------
__global__ void k_embed_nodes(const int* __restrict__ tok, const float* __restrict__ emb,
                              float* __restrict__ h, int N) {
    long gid = (long)blockIdx.x * blockDim.x + threadIdx.x;
    if (gid >= (long)N * D32) return;
    int row = (int)(gid >> 5), lane = (int)(gid & 31);
    h[gid] = relu_(emb[tok[row] * D32 + lane]);
}
__global__ void k_embed_edges(const int* __restrict__ tok, const float* __restrict__ emb,
                              float* __restrict__ e, long E) {
    long gid = (long)blockIdx.x * blockDim.x + threadIdx.x;
    if (gid >= E * D32) return;
    int row = (int)(gid >> 5), lane = (int)(gid & 31);
    e[gid] = emb[tok[row] * D32 + lane];
}

// ---------------------------------------------------------------------------
// Kernel A: node GEMMs  hi = h@Wni, hj = h@Wnj, hn = h@Wnode  (N x 32x32)
// Also zero-inits this layer's softmax accumulators (m, denom, haccum).
// One wave -> 16 nodes (16x32 tile), 6 WMMAs.
// ---------------------------------------------------------------------------
__global__ void k_node_gemm(const float* __restrict__ h,
                            const float* __restrict__ Wni,
                            const float* __restrict__ Wnj,
                            const float* __restrict__ Wnode,
                            float* __restrict__ hi, float* __restrict__ hj,
                            float* __restrict__ hn,
                            float* __restrict__ haccum, float* __restrict__ denom,
                            unsigned* __restrict__ mmax, int N) {
    __shared__ float lw[3 * D32 * D32];
    int tid = threadIdx.x;
    for (int i = tid; i < 1024; i += 256) {
        lw[i] = Wni[i]; lw[1024 + i] = Wnj[i]; lw[2048 + i] = Wnode[i];
    }
    __syncthreads();

    int wave = tid >> 5, lane = tid & 31;
    int tile = blockIdx.x * 8 + wave;
    if (tile * 16 >= N) return;                 // wave-uniform exit; tiles are full
    int row0 = tile * 16;
    bool hi16 = lane >= 16;
    int m_ = lane & 15, n_ = lane & 15;
    int kbase = hi16 ? 8 : 0;                   // unused; A frag also uses kb2 (k = 16h + s)
    int kb2   = hi16 ? 16 : 0;                  // B frag K-base
    int rbase = row0 + (hi16 ? 8 : 0);          // C frag: VGPR r -> row rbase + r

    const float* rp = h + (size_t)(row0 + m_) * D32;
    v16h a;
    #pragma unroll
    for (int t = 0; t < 8; ++t) {
        a[t]     = (_Float16)rp[kb2 + t];
        a[8 + t] = (_Float16)rp[kb2 + 8 + t];
    }

    #pragma unroll
    for (int wsel = 0; wsel < 3; ++wsel) {
        const float* W = lw + wsel * 1024;
        float* outp = (wsel == 0) ? hi : ((wsel == 1) ? hj : hn);
        #pragma unroll
        for (int nt = 0; nt < 2; ++nt) {
            v16h b;
            #pragma unroll
            for (int t = 0; t < 16; ++t)
                b[t] = (_Float16)W[(kb2 + t) * D32 + nt * 16 + n_];
            v8f c = {};
            c = __builtin_amdgcn_wmma_f32_16x16x32_f16(false, a, false, b,
                                                       (short)0, c, false, false);
            float* op = outp + (size_t)rbase * D32 + nt * 16 + n_;
            #pragma unroll
            for (int r = 0; r < 8; ++r) op[r * D32] = c[r];   // clause stores
        }
    }
    // init this layer's accumulators for the 16 nodes of this tile
    if (lane < 16) { denom[row0 + lane] = 0.0f; mmax[row0 + lane] = 0u; }
    float* hp = haccum + (size_t)row0 * D32 + lane;
    #pragma unroll
    for (int r = 0; r < 16; ++r) hp[r * D32] = 0.0f;
}

// ---------------------------------------------------------------------------
// Kernel B: edge GEMM + fused epilogue (IN-PLACE on e):
//   f = leaky_relu(e@Wfij + hi[src] + hj[dst] + b);  e <- f
//   logits = f . attn ;  atomicMax segment-max over dst
// One wave -> 16 edges, 2 WMMAs.
// ---------------------------------------------------------------------------
__global__ void k_edge_gemm(const float* __restrict__ hi, const float* __restrict__ hj,
                            float* __restrict__ e,
                            const int* __restrict__ src, const int* __restrict__ dst,
                            const float* __restrict__ Wf, const float* __restrict__ bvec,
                            const float* __restrict__ attn,
                            float* __restrict__ logits, unsigned* __restrict__ mmax,
                            int E) {
    __shared__ float lw[D32 * D32];
    __shared__ float lb[D32];
    __shared__ float lat[D32];
    int tid = threadIdx.x;
    for (int i = tid; i < 1024; i += 256) lw[i] = Wf[i];
    if (tid < D32) { lb[tid] = bvec[tid]; lat[tid] = attn[tid]; }
    __syncthreads();

    int wave = tid >> 5, lane = tid & 31;
    int tile = blockIdx.x * 8 + wave;
    if (tile * 16 >= E) return;                 // wave-uniform exit; tiles are full
    int row0 = tile * 16;
    bool hi16 = lane >= 16;
    int m_ = lane & 15, n_ = lane & 15;
    int kbase = hi16 ? 8 : 0;
    int kb2   = hi16 ? 16 : 0;
    int rbase = row0 + (hi16 ? 8 : 0);

    const float* rp = e + (size_t)(row0 + m_) * D32;
    v16h a;
    #pragma unroll
    for (int t = 0; t < 8; ++t) {
        a[t]     = (_Float16)rp[kb2 + t];
        a[8 + t] = (_Float16)rp[kb2 + 8 + t];
    }
    v16h b0, b1;
    #pragma unroll
    for (int t = 0; t < 16; ++t) {
        b0[t] = (_Float16)lw[(kb2 + t) * D32 + n_];
        b1[t] = (_Float16)lw[(kb2 + t) * D32 + 16 + n_];
    }
    v8f c0 = {}, c1 = {};
    c0 = __builtin_amdgcn_wmma_f32_16x16x32_f16(false, a, false, b0, (short)0, c0, false, false);
    c1 = __builtin_amdgcn_wmma_f32_16x16x32_f16(false, a, false, b1, (short)0, c1, false, false);

    float la0 = lat[n_], la1 = lat[16 + n_];
    float lb0 = lb[n_],  lb1 = lb[16 + n_];
    float* ep = e + (size_t)rbase * D32 + n_;           // col n_ of row rbase
    const int* sp = src + rbase;
    const int* dp = dst + rbase;

    #pragma unroll
    for (int r = 0; r < 8; ++r) {
        int s = sp[r], d = dp[r];
        float v0 = c0[r] + hi[(size_t)s * D32 + n_]      + hj[(size_t)d * D32 + n_]      + lb0;
        float v1 = c1[r] + hi[(size_t)s * D32 + 16 + n_] + hj[(size_t)d * D32 + 16 + n_] + lb1;
        v0 = lrelu(v0); v1 = lrelu(v1);
        ep[r * D32]      = v0;                        // f overwrites e in place
        ep[r * D32 + 16] = v1;
        float lg = v0 * la0 + v1 * la1;               // logits = f . attn
        lg += __shfl_xor(lg, 1, 16);
        lg += __shfl_xor(lg, 2, 16);
        lg += __shfl_xor(lg, 4, 16);
        lg += __shfl_xor(lg, 8, 16);
        if (n_ == 0) {                                 // lanes 0 and 16: one row each
            logits[rbase + r] = lg;
            atomicMax(&mmax[d], fenc(lg));
        }
    }
}

// ---------------------------------------------------------------------------
// Kernel C: edge scatter  ex = exp(logit - m[dst]);
//           denom[dst]+=ex;  haccum[dst] += hn[src]*ex   (32 lanes per edge)
// ---------------------------------------------------------------------------
__global__ void k_scatter(const float* __restrict__ hn, const float* __restrict__ logits,
                          const unsigned* __restrict__ mmax,
                          const int* __restrict__ src, const int* __restrict__ dst,
                          float* __restrict__ haccum, float* __restrict__ denom, int E) {
    long gid = (long)blockIdx.x * blockDim.x + threadIdx.x;
    int edge = (int)(gid >> 5), lane = (int)(gid & 31);
    if (edge >= E) return;
    int s = src[edge], d = dst[edge];
    float ex = __expf(logits[edge] - fdec(mmax[d]));
    if (lane == 0) atomicAdd(&denom[d], ex);
    atomicAdd(&haccum[(size_t)d * D32 + lane], hn[(size_t)s * D32 + lane] * ex);
}

// ---------------------------------------------------------------------------
// Kernel D: h = relu(haccum / denom)   (softmax-normalized aggregation)
// ---------------------------------------------------------------------------
__global__ void k_norm(const float* __restrict__ haccum, const float* __restrict__ denom,
                       float* __restrict__ h, int N) {
    long gid = (long)blockIdx.x * blockDim.x + threadIdx.x;
    if (gid >= (long)N * D32) return;
    int row = (int)(gid >> 5);
    float dn = denom[row];
    float v = (dn > 0.0f) ? (haccum[gid] / dn) : 0.0f;
    h[gid] = relu_(v);
}

// ---------------------------------------------------------------------------
// Kernel E: per-node bitonic sort of the 32 channels (wave32 == one row).
// ---------------------------------------------------------------------------
__global__ void k_sortrows(const float* __restrict__ h, float* __restrict__ hs,
                           float* __restrict__ rowmax, int N) {
    int wave = threadIdx.x >> 5, lane = threadIdx.x & 31;
    int row = blockIdx.x * 8 + wave;
    if (row >= N) return;
    float v = h[(size_t)row * D32 + lane];
    #pragma unroll
    for (int k = 2; k <= 32; k <<= 1) {
        #pragma unroll
        for (int j = k >> 1; j >= 1; j >>= 1) {
            float p = __shfl_xor(v, j, 32);
            bool up    = (lane & k) == 0;
            bool lower = (lane & j) == 0;
            v = (up == lower) ? fminf(v, p) : fmaxf(v, p);
        }
    }
    hs[(size_t)row * D32 + lane] = v;
    if (lane == 31) rowmax[row] = v;   // hs[:, -1]
}

// ---------------------------------------------------------------------------
// Kernel F: top-K=8 nodes by rowmax, descending, stable (lower index wins ties)
// ---------------------------------------------------------------------------
__global__ void k_topk(const float* __restrict__ rowmax, int* __restrict__ topk, int N) {
    __shared__ float sv[256];
    __shared__ int   si[256];
    __shared__ int   chosen[8];
    int tid = threadIdx.x;
    for (int t = 0; t < 8; ++t) {
        float bv = -3.0e38f; int bi = 0x7fffffff;
        for (int i = tid; i < N; i += 256) {
            bool skip = false;
            for (int c = 0; c < t; ++c) if (chosen[c] == i) skip = true;
            if (skip) continue;
            float v = rowmax[i];
            if (v > bv || (v == bv && i < bi)) { bv = v; bi = i; }
        }
        sv[tid] = bv; si[tid] = bi;
        __syncthreads();
        for (int s = 128; s > 0; s >>= 1) {
            if (tid < s) {
                if (sv[tid + s] > sv[tid] ||
                    (sv[tid + s] == sv[tid] && si[tid + s] < si[tid])) {
                    sv[tid] = sv[tid + s]; si[tid] = si[tid + s];
                }
            }
            __syncthreads();
        }
        if (tid == 0) { chosen[t] = si[0]; topk[t] = si[0]; }
        __syncthreads();
    }
}

// ---------------------------------------------------------------------------
// Kernel G: gather [K*D]=256 pooled features, run the tiny MLP head.
// ---------------------------------------------------------------------------
__global__ void k_mlp(const float* __restrict__ hs, const int* __restrict__ topk,
                      const float* __restrict__ Wlin, const float* __restrict__ blin,
                      const float* __restrict__ W1, const float* __restrict__ b1,
                      const float* __restrict__ W2, const float* __restrict__ b2,
                      const float* __restrict__ Wc, const float* __restrict__ bc,
                      float* __restrict__ out) {
    __shared__ float x[256], y1[D32], y2[D32], y3[D32];
    int tid = threadIdx.x;
    int node = topk[tid >> 5];
    x[tid] = hs[(size_t)node * D32 + (tid & 31)];
    __syncthreads();
    if (tid < D32) {
        float acc = blin[tid];
        for (int k = 0; k < 256; ++k) acc += x[k] * Wlin[k * D32 + tid];
        y1[tid] = relu_(acc);
    }
    __syncthreads();
    if (tid < D32) {
        float acc = b1[tid];
        for (int k = 0; k < D32; ++k) acc += y1[k] * W1[k * D32 + tid];
        y2[tid] = relu_(acc);
    }
    __syncthreads();
    if (tid < D32) {
        float acc = b2[tid];
        for (int k = 0; k < D32; ++k) acc += y2[k] * W2[k * D32 + tid];
        y3[tid] = relu_(acc);
    }
    __syncthreads();
    if (tid < 2) {
        float acc = bc[tid];
        for (int k = 0; k < D32; ++k) acc += y3[k] * Wc[k * 2 + tid];
        out[tid] = acc;
    }
}

// ---------------------------------------------------------------------------
// Host launcher
// ---------------------------------------------------------------------------
extern "C" void kernel_launch(void* const* d_in, const int* in_sizes, int n_in,
                              void* d_out, int out_size, void* d_ws, size_t ws_size,
                              hipStream_t stream) {
    const int*   h_tok     = (const int*)  d_in[0];
    const int*   e_tok     = (const int*)  d_in[1];
    const int*   src       = (const int*)  d_in[2];
    const int*   dst       = (const int*)  d_in[3];
    const float* tok_emb   = (const float*)d_in[4];
    const float* e_tok_emb = (const float*)d_in[5];
    const float* W_ni      = (const float*)d_in[6];
    const float* W_nj      = (const float*)d_in[7];
    const float* W_fij     = (const float*)d_in[8];
    const float* b_edge    = (const float*)d_in[9];
    const float* attn      = (const float*)d_in[10];
    const float* W_node    = (const float*)d_in[11];
    const float* W_lin     = (const float*)d_in[12];
    const float* b_lin     = (const float*)d_in[13];
    const float* W1        = (const float*)d_in[14];
    const float* b1        = (const float*)d_in[15];
    const float* W2        = (const float*)d_in[16];
    const float* b2        = (const float*)d_in[17];
    const float* Wc        = (const float*)d_in[18];
    const float* bc        = (const float*)d_in[19];

    const int N = in_sizes[0];
    const int E = in_sizes[1];
    const int L = 8;

    // --- workspace carving (256B aligned) ---
    char* wp = (char*)d_ws;
    auto carve = [&](size_t bytes) -> void* {
        void* p = (void*)wp;
        wp += (bytes + 255) & ~(size_t)255;
        return p;
    };
    float*    h      = (float*)   carve((size_t)N * D32 * 4);
    float*    hi     = (float*)   carve((size_t)N * D32 * 4);
    float*    hj     = (float*)   carve((size_t)N * D32 * 4);   // reused as hs after layers
    float*    hn     = (float*)   carve((size_t)N * D32 * 4);
    float*    haccum = (float*)   carve((size_t)N * D32 * 4);
    float*    denom  = (float*)   carve((size_t)N * 4);
    unsigned* mmax   = (unsigned*)carve((size_t)N * 4);
    float*    rowmax = (float*)   carve((size_t)N * 4);
    int*      topk   = (int*)     carve(256);
    float*    e      = (float*)   carve((size_t)E * D32 * 4);
    float*    logits = (float*)   carve((size_t)E * 4);
    (void)ws_size; (void)n_in;

    const int thr = 256;
    long nodeElems = (long)N * D32, edgeElems = (long)E * D32;
    int nodeTileBlocks = ((N + 15) / 16 + 7) / 8;   // 8 waves/block, 16 rows/wave
    int edgeTileBlocks = ((E + 15) / 16 + 7) / 8;

    // embeddings
    k_embed_nodes<<<(int)((nodeElems + thr - 1) / thr), thr, 0, stream>>>(h_tok, tok_emb, h, N);
    k_embed_edges<<<(int)((edgeElems + thr - 1) / thr), thr, 0, stream>>>(e_tok, e_tok_emb, e, (long)E);

    // L EGAT layers
    for (int l = 0; l < L; ++l) {
        const float* Wni = W_ni   + (size_t)l * D32 * D32;
        const float* Wnj = W_nj   + (size_t)l * D32 * D32;
        const float* Wf  = W_fij  + (size_t)l * D32 * D32;
        const float* Wn  = W_node + (size_t)l * D32 * D32;
        const float* bl  = b_edge + (size_t)l * D32;
        const float* at  = attn   + (size_t)l * D32;

        k_node_gemm<<<nodeTileBlocks, thr, 0, stream>>>(h, Wni, Wnj, Wn,
                                                        hi, hj, hn, haccum, denom, mmax, N);
        k_edge_gemm<<<edgeTileBlocks, thr, 0, stream>>>(hi, hj, e, src, dst,
                                                        Wf, bl, at, logits, mmax, E);
        k_scatter<<<(int)((edgeElems + thr - 1) / thr), thr, 0, stream>>>(hn, logits, mmax,
                                                                          src, dst, haccum, denom, E);
        k_norm<<<(int)((nodeElems + thr - 1) / thr), thr, 0, stream>>>(haccum, denom, h, N);
    }

    // SortPooling(K=8) + MLP head
    float* hs = hj;  // reuse
    k_sortrows<<<(N + 7) / 8, thr, 0, stream>>>(h, hs, rowmax, N);
    k_topk<<<1, thr, 0, stream>>>(rowmax, topk, N);
    k_mlp<<<1, thr, 0, stream>>>(hs, topk, W_lin, b_lin, W1, b1, W2, b2, Wc, bc, (float*)d_out);
}
